// PGTAttention_85057532330440
// MI455X (gfx1250) — matrix operand order repaired and checked by reading, not compiled, not run on hardware
//
#include <hip/hip_runtime.h>
#include <hip/hip_bf16.h>
#include <cstdint>

// ---------------------------------------------------------------------------
// MI455X (gfx1250) attention block: QKV GEMM + partial RoPE -> flash attention
// -> dense projection.  v_wmma_f32_16x16x32_bf16 everywhere; weight tiles are
// staged into LDS by the Tensor Data Mover with hardware row padding (80 B
// pitch -> conflict-free ds_load_b128 fragment reads), double-buffered against
// the WMMA pipeline.  Weights are pre-transposed and V is stored [head][d][s]
// so every WMMA operand load is a contiguous 16-byte access.
// ---------------------------------------------------------------------------

typedef __attribute__((ext_vector_type(16))) __bf16   v16bf;
typedef __attribute__((ext_vector_type(8)))  float    v8f;
typedef __attribute__((ext_vector_type(4)))  unsigned u32x4;
typedef __attribute__((ext_vector_type(8)))  int      i32x8;
typedef __attribute__((ext_vector_type(4)))  int      i32x4;

#define S_LEN 4096
#define HID   2048
#define NHEAD 16
#define HDIM  128
#define QKV_N 6144

#if defined(__AMDGCN__) && __has_builtin(__builtin_amdgcn_tensor_load_to_lds)
  #define HAS_TDM 1
#else
  #define HAS_TDM 0
#endif
#if defined(__AMDGCN__) && !HAS_TDM
  #warning "PROBE: __builtin_amdgcn_tensor_load_to_lds NOT available - fallback staging in use"
#endif
#if defined(__AMDGCN__) && __has_builtin(__builtin_amdgcn_s_wait_tensorcnt)
  #define WAIT_TENSORCNT(n) __builtin_amdgcn_s_wait_tensorcnt(n)
#else
  #define WAIT_TENSORCNT(n)
  #if defined(__AMDGCN__)
    #warning "PROBE: __builtin_amdgcn_s_wait_tensorcnt NOT available"
  #endif
#endif

// LDS row pitch for staged tiles: 32 bf16 payload + 8 bf16 pad = 80 bytes.
// 80 B = 20 banks -> the 16 lanes of a fragment read start 4 banks apart:
// 0,20,40,60,16,36,56,12,... all distinct, so b128 reads are conflict-free.
#define BPITCH 40

union BF16Frag {
    v16bf v;
    __bf16 e[16];
    uint4  q[2];
};

__device__ __forceinline__ __bf16 f2bf(float f) {
    union { float f; unsigned u; } a; a.f = f;
    unsigned u = a.u;
    u += 0x7FFFu + ((u >> 16) & 1u);          // round-to-nearest-even
    union { unsigned short s; __bf16 b; } o;
    o.s = (unsigned short)(u >> 16);
    return o.b;
}

// Low 32 bits of a generic LDS pointer == byte offset within the LDS aperture.
__device__ __forceinline__ unsigned lds_offset(const void* p) {
    return (unsigned)(uintptr_t)p;
}

#if HAS_TDM
// 2-D bf16 tile DMA: tile_d0 contiguous elements per row, tile_d1 rows, row
// pitch stride_d0 elements.  With padded=true the TDM inserts 16 B of LDS pad
// after every 64 B stored (pad_interval=16 dwords, pad_amount=4 dwords) ->
// 80 B LDS row pitch.  Tensor extents declared == tile extents (never OOB).
__device__ __forceinline__ void tdm_load_2d_bf16(
    unsigned lds_byte_off, const void* gptr,
    unsigned tile_d0, unsigned tile_d1, unsigned long long stride_d0,
    bool padded)
{
    unsigned long long ga = (unsigned long long)(uintptr_t)gptr;
    unsigned ga_lo = __builtin_amdgcn_readfirstlane((unsigned)(ga & 0xFFFFFFFFull));
    unsigned ga_hi = __builtin_amdgcn_readfirstlane((unsigned)(ga >> 32));
    unsigned loff  = __builtin_amdgcn_readfirstlane(lds_byte_off);

    u32x4 g0;
    g0[0] = 1u;                                        // count=1, user descriptor
    g0[1] = loff;                                      // lds_addr
    g0[2] = ga_lo;                                     // global_addr[31:0]
    g0[3] = (ga_hi & 0x01FFFFFFu) | (2u << 30);        // global_addr[56:32], type=2

    unsigned flags = (1u << 16);                       // data_size=1 -> 2 bytes
    if (padded)
        flags |= (1u << 20) | (3u << 22) | (3u << 25); // pad_en, 16 dw, 4 dw

    i32x8 g1;
    g1[0] = (int)flags;
    g1[1] = (int)((tile_d0 & 0xFFFFu) << 16);          // tensor_dim0[15:0] @ b63:48
    g1[2] = (int)(((tile_d0 >> 16) & 0xFFFFu) |        // tensor_dim0[31:16]
                  ((tile_d1 & 0xFFFFu) << 16));        // tensor_dim1[15:0]
    g1[3] = (int)(((tile_d1 >> 16) & 0xFFFFu) |        // tensor_dim1[31:16]
                  ((tile_d0 & 0xFFFFu) << 16));        // tile_dim0
    g1[4] = (int)(tile_d1 & 0xFFFFu);                  // tile_dim1, tile_dim2=0
    g1[5] = (int)(unsigned)(stride_d0 & 0xFFFFFFFFull);       // dim0 stride lo
    g1[6] = (int)(unsigned)((stride_d0 >> 32) & 0xFFFFull);   // dim0 stride hi
    g1[7] = 0;

    i32x4 g2; g2[0] = 1; g2[1] = 1; g2[2] = 0; g2[3] = 0;     // tensor_dim2/3 = 1
    i32x4 g3; g3[0] = 0; g3[1] = (int)(1u << 16); g3[2] = 0; g3[3] = 0; // tensor_dim4=1
#if __has_include(<hip/amd_detail/amd_gfx1250_TDM.h>)
    i32x8 g4 = {};
    __builtin_amdgcn_tensor_load_to_lds(g0, g1, g2, g3, g4, 0);   // 6-arg toolchain
#else
    __builtin_amdgcn_tensor_load_to_lds(g0, g1, g2, g3, 0);       // 5-arg (ROCm 7.2)
#endif
}
#endif

// ------------------------------ fp32 -> bf16 -------------------------------
__global__ void cvt_f32_bf16(const float* __restrict__ in,
                             __bf16* __restrict__ out, int n) {
    int i = blockIdx.x * blockDim.x + threadIdx.x;
    if (i < n) out[i] = f2bf(in[i]);
}

// ----------------- fp32 [R][C] -> bf16 [C][R] tiled transpose ---------------
__global__ __launch_bounds__(256)
void transpose_cvt(const float* __restrict__ in, __bf16* __restrict__ out,
                   int R, int C) {
    __shared__ float tile[32][33];
    const int c0 = blockIdx.x * 32, r0 = blockIdx.y * 32;
    const int tx = threadIdx.x;
    for (int i = threadIdx.y; i < 32; i += 8)
        tile[i][tx] = in[(size_t)(r0 + i) * C + c0 + tx];
    __syncthreads();
    for (int i = threadIdx.y; i < 32; i += 8)
        out[(size_t)(c0 + i) * R + r0 + tx] = f2bf(tile[tx][i]);
}

// ----------------------- QKV GEMM + bias + RoPE + scatter ------------------
// Block: 256 thr = 8 waves.  Wave tile: 16 (M) x 64 (N).  Grid: (6144/64, 4096/128).
// Weight tile (64 n-rows x 32 k) staged via padded TDM, double-buffered.
__global__ __launch_bounds__(256)
void qkv_rope_kernel(const __bf16* __restrict__ X, const __bf16* __restrict__ Wt,
                     const float* __restrict__ bias,
                     __bf16* __restrict__ Q, __bf16* __restrict__ K,
                     __bf16* __restrict__ Vt) {
    __shared__ __align__(16) unsigned short btile[2][64 * BPITCH];   // 10 KB
    const int lane = threadIdx.x & 31;
    const int wv   = threadIdx.x >> 5;
    const int l15  = lane & 15;
    const int g    = lane >> 4;
    const int M0 = blockIdx.y * 128 + wv * 16;
    const int N0 = blockIdx.x * 64;

    v8f acc[4] = {};
    const __bf16* arow = X + (size_t)(M0 + l15) * HID;

#if HAS_TDM
    if (wv == 0)
        tdm_load_2d_bf16(lds_offset(&btile[0][0]),
                         Wt + (size_t)N0 * HID, 32, 64, HID, true);
#else
    {
        const int row = threadIdx.x >> 2, chunk = threadIdx.x & 3;
        uint4 d = *(const uint4*)(Wt + (size_t)(N0 + row) * HID + chunk * 8);
        *(uint4*)(&btile[0][row * BPITCH + chunk * 8]) = d;
    }
#endif

    for (int kk = 0; kk < HID; kk += 32) {
        const int  cur = (kk >> 5) & 1;
        const bool nxt = (kk + 32) < HID;
#if HAS_TDM
        if (wv == 0) {
            if (nxt)
                tdm_load_2d_bf16(lds_offset(&btile[cur ^ 1][0]),
                                 Wt + (size_t)N0 * HID + kk + 32, 32, 64, HID, true);
            if (nxt) WAIT_TENSORCNT(1); else WAIT_TENSORCNT(0);
        }
#else
        if (nxt) {
            const int row = threadIdx.x >> 2, chunk = threadIdx.x & 3;
            uint4 d = *(const uint4*)(Wt + (size_t)(N0 + row) * HID + kk + 32 + chunk * 8);
            *(uint4*)(&btile[cur ^ 1][row * BPITCH + chunk * 8]) = d;
        }
#endif
        __syncthreads();                       // staged tile visible to all waves

        if (nxt) __builtin_prefetch((const void*)(arow + kk + 32), 0, 1);
        BF16Frag a;
        a.q[0] = *(const uint4*)(arow + kk + g * 16);
        a.q[1] = *(const uint4*)(arow + kk + g * 16 + 8);
        #pragma unroll
        for (int t = 0; t < 4; ++t) {
            BF16Frag b;                        // 2 x ds_load_b128, conflict-free
            const unsigned short* bp = &btile[cur][(t * 16 + l15) * BPITCH + g * 16];
            b.q[0] = *(const uint4*)(bp);
            b.q[1] = *(const uint4*)(bp + 8);
            acc[t] = __builtin_amdgcn_wmma_f32_16x16x32_bf16(
                false, a.v, false, b.v, (short)0, acc[t], false, false);
        }
        __syncthreads();                       // reads done before buffer reuse
    }

    // Epilogue: bias, RoPE (tiles 0/1 hold the (d, d+16) pair when the wave's
    // 64-col window starts a q- or k-section), scatter Q/K [h][s][d], V [h][d][s].
    const int  w384     = N0 % 384;
    const bool ropePair = (w384 == 0) || (w384 == 128);
    const float inv_freq = __powf(10000.0f, -(float)(2 * l15) / 32.0f);

    int   nidx[4];
    float bl[4];
    #pragma unroll
    for (int t = 0; t < 4; ++t) {
        nidx[t] = N0 + t * 16 + l15;
        bl[t]   = bias[nidx[t]];
    }

    #pragma unroll
    for (int r = 0; r < 8; ++r) {
        const int spos = M0 + r + (g ? 8 : 0);
        float x[4];
        #pragma unroll
        for (int t = 0; t < 4; ++t) x[t] = acc[t][r] + bl[t];
        if (ropePair) {
            const float ang = (float)spos * inv_freq;
            const float c = __cosf(ang), sn = __sinf(ang);
            const float t0 = x[0] * c - x[1] * sn;
            const float t1 = x[1] * c + x[0] * sn;
            x[0] = t0; x[1] = t1;
        }
        #pragma unroll
        for (int t = 0; t < 4; ++t) {
            const int n    = nidx[t];
            const int head = n / 384;
            const int rem  = n % 384;
            const int sec  = rem >> 7;
            const int d    = rem & 127;
            const __bf16 bv = f2bf(x[t]);
            if (sec == 0)
                Q[((size_t)head * S_LEN + spos) * HDIM + d] = bv;
            else if (sec == 1)
                K[((size_t)head * S_LEN + spos) * HDIM + d] = bv;
            else  // V stored transposed: [head][d][s]
                Vt[((size_t)head * HDIM + d) * S_LEN + spos] = bv;
        }
    }
}

// ----------------------------- Flash attention -----------------------------
// Block: 256 thr = 8 waves; each wave owns 16 queries of one head.
// Q/K/V fragment loads are all contiguous b128; P transposed through padded LDS.
__global__ __launch_bounds__(256)
void attn_kernel(const __bf16* __restrict__ Qg, const __bf16* __restrict__ Kg,
                 const __bf16* __restrict__ Vtg, const float* __restrict__ amask,
                 __bf16* __restrict__ Og) {
    __shared__ __align__(16) unsigned short plds[8 * 16 * BPITCH];  // 10 KB
    const int lane = threadIdx.x & 31;
    const int wv   = threadIdx.x >> 5;
    const int l15  = lane & 15;
    const int g    = lane >> 4;
    const int head = blockIdx.y;
    const int q0   = blockIdx.x * 128 + wv * 16;
    unsigned short* pl = plds + wv * 16 * BPITCH;

    const __bf16* Qb  = Qg  + (size_t)head * S_LEN * HDIM;
    const __bf16* Kb  = Kg  + (size_t)head * S_LEN * HDIM;
    const __bf16* Vtb = Vtg + (size_t)head * HDIM * S_LEN;

    // Preload all four 16x32 Q fragments (K = 128 dims).
    BF16Frag qa[4];
    {
        const __bf16* qrow = Qb + (size_t)(q0 + l15) * HDIM;
        #pragma unroll
        for (int db = 0; db < 4; ++db) {
            qa[db].q[0] = *(const uint4*)(qrow + db * 32 + g * 16);
            qa[db].q[1] = *(const uint4*)(qrow + db * 32 + g * 16 + 8);
        }
    }

    v8f   accO[8] = {};
    float mrow[8], lrow[8];
    #pragma unroll
    for (int r = 0; r < 8; ++r) { mrow[r] = -1e30f; lrow[r] = 0.0f; }

    const float scale = 0.08838834764831845f;   // 1/sqrt(128)
    const int   nkb   = ((q0 + 15) >> 5) + 1;   // causal: 32-key blocks

    for (int kb = 0; kb < nkb; ++kb) {
        const int key0 = kb * 32;

        // ---- scores: two 16x16 tiles (keys key0..+15, +16..+31) ----
        v8f sacc[2] = {};
        #pragma unroll
        for (int sub = 0; sub < 2; ++sub) {
            const int key = key0 + sub * 16 + l15;
            const __bf16* krow = Kb + (size_t)key * HDIM;
            #pragma unroll
            for (int db = 0; db < 4; ++db) {
                BF16Frag b;              // B = K^T: contiguous 16 dims per lane
                b.q[0] = *(const uint4*)(krow + db * 32 + g * 16);
                b.q[1] = *(const uint4*)(krow + db * 32 + g * 16 + 8);
                sacc[sub] = __builtin_amdgcn_wmma_f32_16x16x32_bf16(
                    false, qa[db].v, false, b.v, (short)0, sacc[sub], false, false);
            }
        }

        // ---- online softmax (row stats via 16-lane shfl reductions) ----
        const float am0 = amask[key0 + l15];
        const float am1 = amask[key0 + 16 + l15];
        float p0[8], p1[8], alpha[8];
        #pragma unroll
        for (int r = 0; r < 8; ++r) {
            const int qr  = q0 + r + (g ? 8 : 0);
            const int k0i = key0 + l15;
            const int k1i = key0 + 16 + l15;
            float s0 = sacc[0][r] * scale + am0;
            float s1 = sacc[1][r] * scale + am1;
            const bool oob0 = (k0i > qr), oob1 = (k1i > qr);
            if (oob0) s0 = -1e30f;
            if (oob1) s1 = -1e30f;
            float mx = fmaxf(s0, s1);
            #pragma unroll
            for (int off = 1; off < 16; off <<= 1)
                mx = fmaxf(mx, __shfl_xor(mx, off, 16));
            const float mn = fmaxf(mrow[r], mx);
            const float a  = __expf(mrow[r] - mn);
            const float e0 = oob0 ? 0.0f : __expf(s0 - mn);
            const float e1 = oob1 ? 0.0f : __expf(s1 - mn);
            float rs = e0 + e1;
            #pragma unroll
            for (int off = 1; off < 16; off <<= 1)
                rs += __shfl_xor(rs, off, 16);
            lrow[r]  = lrow[r] * a + rs;
            mrow[r]  = mn;
            alpha[r] = a;
            p0[r] = e0; p1[r] = e1;
        }
        #pragma unroll
        for (int t = 0; t < 8; ++t)
            #pragma unroll
            for (int r = 0; r < 8; ++r)
                accO[t][r] *= alpha[r];

        // ---- P tile (16q x 32k): C/D layout -> padded LDS -> A layout ----
        #pragma unroll
        for (int r = 0; r < 8; ++r) {
            const int row = r + (g ? 8 : 0);
            union { __bf16 b; unsigned short s; } c0, c1;
            c0.b = f2bf(p0[r]); c1.b = f2bf(p1[r]);
            pl[row * BPITCH + l15]      = c0.s;
            pl[row * BPITCH + 16 + l15] = c1.s;
        }
        __builtin_amdgcn_wave_barrier();
        asm volatile("" ::: "memory");   // same-wave LDS: hw in-order, fence compiler

        BF16Frag pa;                     // 2 x ds_load_b128, conflict-free
        pa.q[0] = *(const uint4*)(pl + l15 * BPITCH + g * 16);
        pa.q[1] = *(const uint4*)(pl + l15 * BPITCH + g * 16 + 8);
        __builtin_amdgcn_wave_barrier();
        asm volatile("" ::: "memory");

        // ---- O += P @ V, V read contiguously from [d][s] layout ----
        #pragma unroll
        for (int t = 0; t < 8; ++t) {
            BF16Frag vb;
            const __bf16* vp = Vtb + (size_t)(t * 16 + l15) * S_LEN + key0 + g * 16;
            vb.q[0] = *(const uint4*)(vp);
            vb.q[1] = *(const uint4*)(vp + 8);
            accO[t] = __builtin_amdgcn_wmma_f32_16x16x32_bf16(
                false, pa.v, false, vb.v, (short)0, accO[t], false, false);
        }
    }

    // ---- normalize and write [S, H] bf16 for the dense GEMM ----
    #pragma unroll
    for (int r = 0; r < 8; ++r) {
        const int spos = q0 + r + (g ? 8 : 0);
        const float inv = 1.0f / lrow[r];
        #pragma unroll
        for (int t = 0; t < 8; ++t) {
            const int col = head * HDIM + t * 16 + l15;
            Og[(size_t)spos * HID + col] = f2bf(accO[t][r] * inv);
        }
    }
}

// ------------------------------- dense GEMM --------------------------------
// attn[4096x2048] @ Wd^T-staged + b -> fp32 out.  Same padded-TDM pipeline.
__global__ __launch_bounds__(256)
void dense_kernel(const __bf16* __restrict__ A, const __bf16* __restrict__ Wt,
                  const float* __restrict__ bias, float* __restrict__ out) {
    __shared__ __align__(16) unsigned short btile[2][64 * BPITCH];   // 10 KB
    const int lane = threadIdx.x & 31;
    const int wv   = threadIdx.x >> 5;
    const int l15  = lane & 15;
    const int g    = lane >> 4;
    const int M0 = blockIdx.y * 128 + wv * 16;
    const int N0 = blockIdx.x * 64;

    v8f acc[4] = {};
    const __bf16* arow = A + (size_t)(M0 + l15) * HID;

#if HAS_TDM
    if (wv == 0)
        tdm_load_2d_bf16(lds_offset(&btile[0][0]),
                         Wt + (size_t)N0 * HID, 32, 64, HID, true);
#else
    {
        const int row = threadIdx.x >> 2, chunk = threadIdx.x & 3;
        uint4 d = *(const uint4*)(Wt + (size_t)(N0 + row) * HID + chunk * 8);
        *(uint4*)(&btile[0][row * BPITCH + chunk * 8]) = d;
    }
#endif

    for (int kk = 0; kk < HID; kk += 32) {
        const int  cur = (kk >> 5) & 1;
        const bool nxt = (kk + 32) < HID;
#if HAS_TDM
        if (wv == 0) {
            if (nxt)
                tdm_load_2d_bf16(lds_offset(&btile[cur ^ 1][0]),
                                 Wt + (size_t)N0 * HID + kk + 32, 32, 64, HID, true);
            if (nxt) WAIT_TENSORCNT(1); else WAIT_TENSORCNT(0);
        }
#else
        if (nxt) {
            const int row = threadIdx.x >> 2, chunk = threadIdx.x & 3;
            uint4 d = *(const uint4*)(Wt + (size_t)(N0 + row) * HID + kk + 32 + chunk * 8);
            *(uint4*)(&btile[cur ^ 1][row * BPITCH + chunk * 8]) = d;
        }
#endif
        __syncthreads();

        if (nxt) __builtin_prefetch((const void*)(arow + kk + 32), 0, 1);
        BF16Frag a;
        a.q[0] = *(const uint4*)(arow + kk + g * 16);
        a.q[1] = *(const uint4*)(arow + kk + g * 16 + 8);
        #pragma unroll
        for (int t = 0; t < 4; ++t) {
            BF16Frag b;
            const unsigned short* bp = &btile[cur][(t * 16 + l15) * BPITCH + g * 16];
            b.q[0] = *(const uint4*)(bp);
            b.q[1] = *(const uint4*)(bp + 8);
            acc[t] = __builtin_amdgcn_wmma_f32_16x16x32_bf16(
                false, a.v, false, b.v, (short)0, acc[t], false, false);
        }
        __syncthreads();
    }

    #pragma unroll
    for (int r = 0; r < 8; ++r) {
        const int spos = M0 + r + (g ? 8 : 0);
        #pragma unroll
        for (int t = 0; t < 4; ++t) {
            const int n = N0 + t * 16 + l15;
            out[(size_t)spos * HID + n] = acc[t][r] + bias[n];
        }
    }
}

// ------------------------------ host launcher ------------------------------
extern "C" void kernel_launch(void* const* d_in, const int* in_sizes, int n_in,
                              void* d_out, int out_size, void* d_ws, size_t ws_size,
                              hipStream_t stream) {
    const float* hs    = (const float*)d_in[0];   // [1,4096,2048]
    const float* amask = (const float*)d_in[1];   // [1,1,1,4096]
    const float* Wqkv  = (const float*)d_in[2];   // [2048,6144]
    const float* bqkv  = (const float*)d_in[3];   // [6144]
    const float* Wd    = (const float*)d_in[4];   // [2048,2048]
    const float* bd    = (const float*)d_in[5];   // [2048]
    float* out = (float*)d_out;

    char* ws = (char*)d_ws;
    const size_t MB = 1024ull * 1024ull;
    __bf16* Xbf    = (__bf16*)(ws);               // 16 MB  [4096][2048]
    __bf16* Wqkvt  = (__bf16*)(ws + 16 * MB);     // 24 MB  [6144][2048] (transposed)
    __bf16* Wdt    = (__bf16*)(ws + 40 * MB);     //  8 MB  [2048][2048] (transposed)
    __bf16* Qb     = (__bf16*)(ws + 48 * MB);     // 16 MB  [16][4096][128]
    __bf16* Kb     = (__bf16*)(ws + 64 * MB);     // 16 MB  [16][4096][128]
    __bf16* Vtb    = (__bf16*)(ws + 80 * MB);     // 16 MB  [16][128][4096]
    __bf16* Ab     = (__bf16*)(ws + 96 * MB);     // 16 MB  [4096][2048]

    const int nX = S_LEN * HID;
    cvt_f32_bf16<<<(nX + 255) / 256, 256, 0, stream>>>(hs, Xbf, nX);
    transpose_cvt<<<dim3(QKV_N / 32, HID / 32), dim3(32, 8), 0, stream>>>(
        Wqkv, Wqkvt, HID, QKV_N);
    transpose_cvt<<<dim3(HID / 32, HID / 32), dim3(32, 8), 0, stream>>>(
        Wd, Wdt, HID, HID);

    qkv_rope_kernel<<<dim3(QKV_N / 64, S_LEN / 128), 256, 0, stream>>>(
        Xbf, Wqkvt, bqkv, Qb, Kb, Vtb);

    attn_kernel<<<dim3(S_LEN / 128, NHEAD), 256, 0, stream>>>(
        Qb, Kb, Vtb, amask, Ab);

    dense_kernel<<<dim3(HID / 64, S_LEN / 128), 256, 0, stream>>>(
        Ab, Wdt, bd, out);
}
